// Reservoir_31001073943126
// MI455X (gfx1250) — compile-verified, matrix operand order checked by reading, not run on hardware
//
#include <hip/hip_runtime.h>
#include <hip/hip_bf16.h>
#include <math.h>
#include <stdint.h>

// Problem constants (from reference)
static constexpr int P_ = 16, N_ = 1024, I_ = 128, B_ = 32, T_ = 64;
static constexpr float ALPHA  = 0.1f;                   // clip(DT/TAU_M,...)
static constexpr float DECAY  = 1.0f - 0.001f * 0.01f;  // 1 - LR*e3
static constexpr float OUTERC = 0.001f * 0.1f;          // LR*e0

typedef __attribute__((ext_vector_type(16))) __bf16 v16bf;
typedef __attribute__((ext_vector_type(8)))  float  v8f;

// Load the 16 fp32 values a lane contributes to one 16x32 bf16 fragment:
// two contiguous 8-float runs at base+0 and base+16 (base is 32B aligned).
__device__ __forceinline__ void load16(const float* __restrict__ base, float v[16]) {
  const float4* q = reinterpret_cast<const float4*>(base);
  float4 a = q[0], b = q[1], c = q[4], d = q[5];
  v[0]=a.x; v[1]=a.y; v[2]=a.z;  v[3]=a.w;
  v[4]=b.x; v[5]=b.y; v[6]=b.z;  v[7]=b.w;
  v[8]=c.x; v[9]=c.y; v[10]=c.z; v[11]=c.w;
  v[12]=d.x; v[13]=d.y; v[14]=d.z; v[15]=d.w;
}

// bf16x2 split: x ~= hi + lo with hi = bf16(x), lo = bf16(x - hi)
__device__ __forceinline__ void split16(const float v[16], v16bf& hi, v16bf& lo) {
#pragma unroll
  for (int i = 0; i < 16; ++i) {
    __bf16 h = (__bf16)v[i];
    hi[i] = h;
    lo[i] = (__bf16)(v[i] - (float)h);
  }
}

__device__ __forceinline__ v8f mma3(v8f c, const v16bf& ah, const v16bf& al,
                                    const v16bf& bh, const v16bf& bl) {
  c = __builtin_amdgcn_wmma_f32_16x16x32_bf16(false, ah, false, bh, (short)0, c, false, false);
  c = __builtin_amdgcn_wmma_f32_16x16x32_bf16(false, ah, false, bl, (short)0, c, false, false);
  c = __builtin_amdgcn_wmma_f32_16x16x32_bf16(false, al, false, bh, (short)0, c, false, false);
  return c;
}

// ---------------- one-time prep kernels ----------------
__global__ void copy4_kernel(float4* __restrict__ dst, const float4* __restrict__ src, int n4) {
  int i = blockIdx.x * blockDim.x + threadIdx.x;
  if (i < n4) dst[i] = src[i];
}

__global__ void maskbits_kernel(uint32_t* __restrict__ dst, const float* __restrict__ mask, int nwords) {
  int i = blockIdx.x * blockDim.x + threadIdx.x;
  if (i < nwords) {
    const float* s = mask + (size_t)i * 32;
    uint32_t b = 0;
#pragma unroll
    for (int k = 0; k < 32; ++k) b |= (s[k] != 0.f) ? (1u << k) : 0u;
    dst[i] = b;
  }
}

__global__ void zero_kernel(float* __restrict__ dst, int n) {
  int i = blockIdx.x * blockDim.x + threadIdx.x;
  if (i < n) dst[i] = 0.f;
}

// ---------------- fused per-timestep kernel ----------------
// grid = P * (N/64) blocks, 256 threads (8 wave32).
// Block (p, m0..m0+63):
//   Phase 1 (t>0): W[p, m0:m0+64, :] <- W*DECAY + maskbit * OUTERC*xc[m]*xr[n]
//   Phase 2: D(32x64) = h_{t-1}(32x1024) @ W[p]^T + x_t @ W_in[p]^T  (bf16x2 WMMA)
//            h_t = 0.9*h_{t-1} + 0.1*tanh(D + bias); write to out[t]; emit xbar[t]
__global__ __launch_bounds__(256) void step_kernel(
    const float* __restrict__ x_seq, const float* __restrict__ W_in,
    const float* __restrict__ bias, float* __restrict__ W,
    const uint32_t* __restrict__ mbits, float* __restrict__ xbar,
    float* __restrict__ out, int t)
{
  const int p   = blockIdx.x >> 4;
  const int m0  = (blockIdx.x & 15) << 6;
  const int tid = threadIdx.x;

  // ---- Phase 1: plasticity update (block-local slab of W) ----
  if (t > 0) {
    const float* xc = xbar + (size_t)(t + 1) * (P_ * N_) + p * N_ + m0; // xpost(t-1)
    const float* xr = xbar + (size_t)(t)     * (P_ * N_) + p * N_;      // xpre(t-1)=xpost(t-2)
    const int ml   = tid >> 2;          // 0..63 : row within slab
    const int nseg = (tid & 3) << 8;    // 4 threads x 256 cols
    const float xcm = OUTERC * xc[ml];
    const size_t rowoff = ((size_t)(p * N_ + m0 + ml)) * N_;
    float* wr = W + rowoff + nseg;
    const uint32_t* mb = mbits + (rowoff >> 5) + (nseg >> 5);
    const float* xrp = xr + nseg;
#pragma unroll 1
    for (int g = 0; g < 8; ++g) {       // 8 mask words of 32 cols
      uint32_t bits = mb[g];
#pragma unroll
      for (int q = 0; q < 8; ++q) {     // 8 float4 per word
        const int n = (g << 5) + (q << 2);
        float4 w = *reinterpret_cast<float4*>(wr + n);
        float4 x = *reinterpret_cast<const float4*>(xrp + n);
        w.x = w.x * DECAY + (((bits >> (q * 4 + 0)) & 1u) ? xcm * x.x : 0.f);
        w.y = w.y * DECAY + (((bits >> (q * 4 + 1)) & 1u) ? xcm * x.y : 0.f);
        w.z = w.z * DECAY + (((bits >> (q * 4 + 2)) & 1u) ? xcm * x.z : 0.f);
        w.w = w.w * DECAY + (((bits >> (q * 4 + 3)) & 1u) ? xcm * x.w : 0.f);
        *reinterpret_cast<float4*>(wr + n) = w;
      }
    }
    __threadfence();
  }
  __syncthreads();

  // ---- Phase 2: WMMA GEMM ----
  const int lane = tid & 31;
  const int wave = tid >> 5;       // 8 waves
  const int bt   = wave >> 2;      // batch tile 0..1 (16 rows each)
  const int mt   = wave & 3;       // m tile 0..3 within slab
  const int row  = lane & 15;      // A-row (batch) == B-col (m) index
  const int kb   = (lane >> 4) << 3; // 0 or 8: k-pair base per ISA layout

  const int m_col = m0 + (mt << 4) + row;
  const float* wrow  = W    + ((size_t)(p * N_ + m_col)) * N_;
  const float* wirow = W_in + ((size_t)(p * N_ + m_col)) * I_;
  const float* xrow  = x_seq + ((size_t)t * B_ + (bt * 16 + row)) * I_;
  const float* hrow  = (t > 0)
      ? out + (((size_t)(t - 1) * P_ + p) * B_ + (bt * 16 + row)) * N_
      : nullptr;

  v8f c = {};
  float av[16], bv[16];
  v16bf ah, al, bh, bl;

  if (t > 0) {
#pragma unroll 2
    for (int n0 = 0; n0 < N_; n0 += 32) {
      __builtin_prefetch(wrow + n0 + 256, 0, 0);  // global_prefetch_b8 on W stream
      load16(hrow + n0 + kb, av);
      load16(wrow + n0 + kb, bv);
      split16(av, ah, al);
      split16(bv, bh, bl);
      c = mma3(c, ah, al, bh, bl);
    }
  }
#pragma unroll
  for (int i0 = 0; i0 < I_; i0 += 32) {
    load16(xrow + i0 + kb, av);
    load16(wirow + i0 + kb, bv);
    split16(av, ah, al);
    split16(bv, bh, bl);
    c = mma3(c, ah, al, bh, bl);
  }

  // ---- Epilogue: leaky tanh update, write states, batch-mean partials ----
  const float bval = bias[p * N_ + m_col];
  float bsum = 0.f;
#pragma unroll
  for (int r = 0; r < 8; ++r) {
    const int b = (bt << 4) + ((lane < 16) ? r : r + 8);  // C layout: VGPR r -> row r / r+8
    const float pre = c[r] + bval;
    float hp = 0.f;
    if (t > 0) hp = out[(((size_t)(t - 1) * P_ + p) * B_ + b) * N_ + m_col];
    const float hn = (1.f - ALPHA) * hp + ALPHA * tanhf(pre);
    out[(((size_t)t * P_ + p) * B_ + b) * N_ + m_col] = hn;
    bsum += hn;
  }
  // reduce the two half-wave batch groups (lanes L and L^16 share the same m)
  bsum += __shfl_xor(bsum, 16, 32);

  __shared__ float smem[128];
  if (lane < 16) smem[bt * 64 + (mt << 4) + lane] = bsum;
  __syncthreads();
  if (tid < 64) {
    const float s = (smem[tid] + smem[64 + tid]) * (1.f / 32.f);
    xbar[(size_t)(t + 2) * (P_ * N_) + p * N_ + m0 + tid] = s;  // slot(t) = t+2
  }
}

// ---------------- host side ----------------
extern "C" void kernel_launch(void* const* d_in, const int* in_sizes, int n_in,
                              void* d_out, int out_size, void* d_ws, size_t ws_size,
                              hipStream_t stream) {
  (void)in_sizes; (void)n_in; (void)out_size; (void)ws_size;
  const float* x_seq = (const float*)d_in[0];
  const float* W_rec = (const float*)d_in[1];
  const float* W_in  = (const float*)d_in[2];
  const float* bias  = (const float*)d_in[3];
  const float* mask  = (const float*)d_in[4];
  float* out = (float*)d_out;

  // Workspace layout: [W_work fp32][mask bitfield][xbar ring: (T+2) slots of P*N]
  const size_t PNN      = (size_t)P_ * N_ * N_;          // 16.8M elems
  char* ws = (char*)d_ws;
  float*    W     = (float*)ws;
  uint32_t* mbits = (uint32_t*)(ws + PNN * sizeof(float));
  float*    xbar  = (float*)(ws + PNN * sizeof(float) + (PNN / 32) * sizeof(uint32_t));

  // Prep: copy W (inputs must not be mutated), compress mask, zero xbar slots -2,-1
  {
    const int n4 = (int)(PNN / 4);
    copy4_kernel<<<(n4 + 255) / 256, 256, 0, stream>>>((float4*)W, (const float4*)W_rec, n4);
    const int nwords = (int)(PNN / 32);
    maskbits_kernel<<<(nwords + 255) / 256, 256, 0, stream>>>(mbits, mask, nwords);
    const int nz = 2 * P_ * N_;
    zero_kernel<<<(nz + 255) / 256, 256, 0, stream>>>(xbar, nz);
  }

  // Sequential recurrence: one fused update+GEMM kernel per timestep
  for (int t = 0; t < T_; ++t) {
    step_kernel<<<dim3(P_ * (N_ / 64)), 256, 0, stream>>>(
        x_seq, W_in, bias, W, mbits, xbar, out, t);
  }
}
